// AttentionLayer_3075196584668
// MI455X (gfx1250) — matrix operand order repaired and checked by reading, not compiled, not run on hardware
//
#include <hip/hip_runtime.h>
#include <hip/hip_bf16.h>
#include <math.h>

#define B_   2
#define S_   2048
#define D_   1024
#define H_   16
#define DH_  64
#define DFF_ 4096

typedef __attribute__((ext_vector_type(16))) __bf16 v16bf;
typedef __attribute__((ext_vector_type(8)))  __bf16 v8bf;
typedef __attribute__((ext_vector_type(8)))  float  v8f;

// ---- CDNA5 async global->LDS copy (guarded probe) --------------------------
#if defined(__has_builtin)
#if __has_builtin(__builtin_amdgcn_global_load_async_to_lds_b128) && \
    __has_builtin(__builtin_amdgcn_s_wait_asynccnt)
#define USE_ASYNC_LDS 1
#endif
#endif

#ifdef USE_ASYNC_LDS
// builtin signature (from probe diagnostics): (v4i AS1* src, v4i AS3* dst, Imm, Imm)
typedef int v4i32 __attribute__((vector_size(16)));
static __device__ __forceinline__ void async_cp16(const __bf16* g, __bf16* l) {
  __builtin_amdgcn_global_load_async_to_lds_b128(
      (__attribute__((address_space(1))) v4i32*)(v4i32*)(void*)g,
      (__attribute__((address_space(3))) v4i32*)(v4i32*)(void*)l,
      0, 0);
}
static __device__ __forceinline__ void async_wait0() {
  __builtin_amdgcn_s_wait_asynccnt(0);
}
#else
static __device__ __forceinline__ void async_wait0() {}
#endif

// ---- fragment helpers (CDNA5 WMMA 16x16x32 bf16 layouts) -------------------
static __device__ __forceinline__ v16bf make_frag(const __bf16* p0, const __bf16* p1) {
  v8bf lo = *(const v8bf*)p0;
  v8bf hi = *(const v8bf*)p1;
  return __builtin_shufflevector(lo, hi, 0,1,2,3,4,5,6,7,8,9,10,11,12,13,14,15);
}
// A-matrix 16x32: lane-half holds K chunks {kh*8..+7} and {16+kh*8..+7} of row m=lane%16
static __device__ __forceinline__ v16bf load_a_frag(const __bf16* rowbase, int kh) {
  return make_frag(rowbase + kh * 8, rowbase + 16 + kh * 8);
}
// B-matrix 32x16: lane-half holds K chunks {kh*8..+7} and {16+kh*8..+7} of column n=lane%16
static __device__ __forceinline__ v16bf load_b_frag(const __bf16* colbase, int kh) {
  return make_frag(colbase + kh * 8, colbase + 16 + kh * 8);
}
static __device__ __forceinline__ v8f wmma_bf16(v16bf a, v16bf b, v8f c) {
  return __builtin_amdgcn_wmma_f32_16x16x32_bf16(false, a, false, b, (short)0, c,
                                                 false, false);
}

// ---- fp32 -> bf16 weight conversion ----------------------------------------
__global__ __launch_bounds__(256) void cvt_bf16_kernel(const float* __restrict__ src,
                                                       __bf16* __restrict__ dst,
                                                       int n4) {
  const int i = (blockIdx.x * 256 + threadIdx.x) * 4;
  if (i < n4) {
    const float4 f = *(const float4*)(src + i);
    __bf16* d = dst + i;
    d[0] = (__bf16)f.x; d[1] = (__bf16)f.y;
    d[2] = (__bf16)f.z; d[3] = (__bf16)f.w;
  }
}

// ---- RMSNorm: x[row,1024] -> bf16 xn ---------------------------------------
__global__ __launch_bounds__(256) void rmsnorm_kernel(const float* __restrict__ x,
                                                      const float* __restrict__ w,
                                                      __bf16* __restrict__ xn) {
  const int row = blockIdx.x;
  const float* xr = x + (size_t)row * D_;
  float ss = 0.f;
  for (int i = threadIdx.x; i < D_; i += 256) { float t = xr[i]; ss += t * t; }
  ss += __shfl_xor(ss, 1);  ss += __shfl_xor(ss, 2);
  ss += __shfl_xor(ss, 4);  ss += __shfl_xor(ss, 8);
  ss += __shfl_xor(ss, 16);
  __shared__ float red[8];
  if ((threadIdx.x & 31) == 0) red[threadIdx.x >> 5] = ss;
  __syncthreads();
  float tot = red[0];
  #pragma unroll
  for (int i = 1; i < 8; ++i) tot += red[i];
  const float r = rsqrtf(tot * (1.f / D_) + 1.1920929e-7f);
  for (int i = threadIdx.x; i < D_; i += 256)
    xn[(size_t)row * D_ + i] = (__bf16)(xr[i] * r * w[i]);
}

// ---- tiled WMMA GEMM: C[M,N] = A[M,K](bf16) @ Wb[N,K](bf16)^T + bias -------
// Double-buffered LDS; async global->LDS staging when available.
// OUTMODE: 0 = bf16 store, 1 = f32 store, 2 = f32 accumulate
template <int OUTMODE>
__global__ __launch_bounds__(256) void gemm_wmma(const __bf16* __restrict__ A,
                                                 const __bf16* __restrict__ Wb,
                                                 const float* __restrict__ bias,
                                                 void* __restrict__ Cout,
                                                 int N, int K) {
  const int tid    = threadIdx.x;
  const int lane   = tid & 31;
  const int wave   = tid >> 5;
  const int lanelo = lane & 15;
  const int kh     = lane >> 4;
  const int wm     = wave >> 2;            // 0..1  (64 rows each)
  const int wn     = wave & 3;             // 0..3  (32 cols each)
  const int m0     = blockIdx.y * 128;
  const int n0     = blockIdx.x * 128;

  __shared__ __bf16 As[2][128 * 40];
  __shared__ __bf16 Bs[2][128 * 40];

  // stage one 128x32 bf16 tile pair into buffer `buf` for K offset `k0`
  auto stage = [&](int buf, int k0) {
    #pragma unroll
    for (int it = 0; it < 2; ++it) {
      const int c   = tid + it * 256;       // 512 chunks of 8 halfs per tile
      const int row = c >> 2;
      const int col = (c & 3) * 8;
      const __bf16* ga = A  + (size_t)(m0 + row) * K + k0 + col;
      const __bf16* gb = Wb + (size_t)(n0 + row) * K + k0 + col;
#ifdef USE_ASYNC_LDS
      async_cp16(ga, &As[buf][row * 40 + col]);
      async_cp16(gb, &Bs[buf][row * 40 + col]);
#else
      *(v8bf*)(&As[buf][row * 40 + col]) = *(const v8bf*)ga;
      *(v8bf*)(&Bs[buf][row * 40 + col]) = *(const v8bf*)gb;
#endif
    }
  };

  v8f zero = {};
  v8f acc[4][2];
  #pragma unroll
  for (int i = 0; i < 4; ++i)
    #pragma unroll
    for (int j = 0; j < 2; ++j) acc[i][j] = zero;

  const int nt = K >> 5;                    // K / 32 steps
  stage(0, 0);
  async_wait0();
  __syncthreads();

  for (int t = 0; t < nt; ++t) {
    const int cur = t & 1;
    if (t + 1 < nt) stage(1 - cur, (t + 1) << 5);

    v16bf af[4], bfr[2];
    #pragma unroll
    for (int i = 0; i < 4; ++i)
      af[i] = load_a_frag(&As[cur][(wm * 64 + i * 16 + lanelo) * 40], kh);
    #pragma unroll
    for (int j = 0; j < 2; ++j)
      bfr[j] = load_b_frag(&Bs[cur][(wn * 32 + j * 16 + lanelo) * 40], kh);
    #pragma unroll
    for (int i = 0; i < 4; ++i)
      #pragma unroll
      for (int j = 0; j < 2; ++j)
        acc[i][j] = wmma_bf16(af[i], bfr[j], acc[i][j]);

    async_wait0();                          // next tile landed in LDS
    __syncthreads();                        // one barrier per K-step
  }

  // epilogue: C VGPR r <-> row r + 8*kh, col = lane%16
  #pragma unroll
  for (int i = 0; i < 4; ++i) {
    #pragma unroll
    for (int j = 0; j < 2; ++j) {
      const int col = n0 + wn * 32 + j * 16 + lanelo;
      const float bv = bias[col];
      #pragma unroll
      for (int r = 0; r < 8; ++r) {
        const int row = m0 + wm * 64 + i * 16 + r + 8 * kh;
        const float val = acc[i][j][r] + bv;
        const size_t o = (size_t)row * N + col;
        if (OUTMODE == 0)      ((__bf16*)Cout)[o] = (__bf16)val;
        else if (OUTMODE == 1) ((float*)Cout)[o] = val;
        else                   ((float*)Cout)[o] += val;
      }
    }
  }
}

// ---- flash attention with ALiBi (full, non-causal) -------------------------
// grid: (S/128, H, B); 8 waves/block, each wave owns 16 query rows
__global__ __launch_bounds__(256) void attention_wmma(const __bf16* __restrict__ Q,
                                                      const __bf16* __restrict__ Kb,
                                                      const __bf16* __restrict__ V,
                                                      __bf16* __restrict__ Ctx) {
  const int tid    = threadIdx.x;
  const int lane   = tid & 31;
  const int wave   = tid >> 5;
  const int lanelo = lane & 15;
  const int kh     = lane >> 4;
  const int h      = blockIdx.y;
  const int b      = blockIdx.z;
  const int q0     = blockIdx.x * 128 + wave * 16;
  const size_t rowbase = (size_t)b * S_;
  const float slope = exp2f(-0.5f * (float)(h + 1));  // 2^(-8(h+1)/H), H=16
  const float scale = 0.125f;                         // 1/sqrt(DH)

  __shared__ __bf16 Vt[64 * 40];        // V tile transposed: [dh][kv]
  __shared__ __bf16 P[8 * 16 * 40];     // per-wave probability tiles
  __bf16* Pw = P + wave * 16 * 40;

  // Q fragments (K = DH = 64 -> 2 A-frags), straight from global
  v16bf qf[2];
  {
    const __bf16* qrow = Q + (rowbase + q0 + lanelo) * D_ + h * DH_;
    qf[0] = load_a_frag(qrow, kh);
    qf[1] = load_a_frag(qrow + 32, kh);
  }

  v8f zero = {};
  v8f ctx[4];
  #pragma unroll
  for (int i = 0; i < 4; ++i) ctx[i] = zero;
  float mrow[8], lrow[8];
  #pragma unroll
  for (int r = 0; r < 8; ++r) { mrow[r] = -1e30f; lrow[r] = 0.f; }

  for (int kv0 = 0; kv0 < S_; kv0 += 32) {
    __syncthreads();                         // previous PV readers done
    {   // stage V tile transposed: thread -> one kv row, 8 dh values
      const int kv  = tid >> 3;
      const int dh0 = (tid & 7) * 8;
      v8bf vv = *(const v8bf*)(V + (rowbase + kv0 + kv) * D_ + h * DH_ + dh0);
      #pragma unroll
      for (int j = 0; j < 8; ++j) Vt[(dh0 + j) * 40 + kv] = vv[j];
    }
    __syncthreads();

    // scores 16x32: two n-tiles, two K-steps over DH
    v8f s0 = zero, s1 = zero;
    {
      const __bf16* kr0 = Kb + (rowbase + kv0 + lanelo) * D_ + h * DH_;
      const __bf16* kr1 = Kb + (rowbase + kv0 + 16 + lanelo) * D_ + h * DH_;
      s0 = wmma_bf16(qf[0], load_b_frag(kr0, kh), s0);
      s0 = wmma_bf16(qf[1], load_b_frag(kr0 + 32, kh), s0);
      s1 = wmma_bf16(qf[0], load_b_frag(kr1, kh), s1);
      s1 = wmma_bf16(qf[1], load_b_frag(kr1 + 32, kh), s1);
    }

    // online softmax with ALiBi; row r of VGPR slot = qpos q0+r+8*kh
    #pragma unroll
    for (int r = 0; r < 8; ++r) {
      const int qpos = q0 + r + 8 * kh;
      const float d0 = fabsf((float)(qpos - (kv0 + lanelo)));
      const float d1 = fabsf((float)(qpos - (kv0 + 16 + lanelo)));
      const float v0 = s0[r] * scale - slope * d0;
      const float v1 = s1[r] * scale - slope * d1;
      float mx = fmaxf(v0, v1);
      mx = fmaxf(mx, __shfl_xor(mx, 1));
      mx = fmaxf(mx, __shfl_xor(mx, 2));
      mx = fmaxf(mx, __shfl_xor(mx, 4));
      mx = fmaxf(mx, __shfl_xor(mx, 8));
      const float mnew = fmaxf(mrow[r], mx);
      const float corr = __expf(mrow[r] - mnew);
      const float p0 = __expf(v0 - mnew);
      const float p1 = __expf(v1 - mnew);
      float ps = p0 + p1;
      ps += __shfl_xor(ps, 1);
      ps += __shfl_xor(ps, 2);
      ps += __shfl_xor(ps, 4);
      ps += __shfl_xor(ps, 8);
      lrow[r] = lrow[r] * corr + ps;
      mrow[r] = mnew;
      #pragma unroll
      for (int i = 0; i < 4; ++i) ctx[i][r] *= corr;
      const int m = r + 8 * kh;
      Pw[m * 40 + lanelo]      = (__bf16)p0;
      Pw[m * 40 + 16 + lanelo] = (__bf16)p1;
    }
    __syncthreads();                         // P visible, Vt stable

    // ctx += P @ V : K = 32 kv, four dh n-tiles
    {
      v16bf pf = load_a_frag(Pw + lanelo * 40, kh);
      #pragma unroll
      for (int i = 0; i < 4; ++i) {
        v16bf vf = load_b_frag(Vt + (i * 16 + lanelo) * 40, kh);
        ctx[i] = wmma_bf16(pf, vf, ctx[i]);
      }
    }
  }

  // normalize and store ctx (bf16)
  #pragma unroll
  for (int i = 0; i < 4; ++i) {
    const int col = h * DH_ + i * 16 + lanelo;
    #pragma unroll
    for (int r = 0; r < 8; ++r) {
      const int row = q0 + r + 8 * kh;
      Ctx[(rowbase + row) * D_ + col] = (__bf16)(ctx[i][r] / lrow[r]);
    }
  }
}

// ---- elementwise: gated = sigmoid(gate) * ctx (in-place into gate) ---------
__global__ void gatemul_kernel(__bf16* __restrict__ g, const __bf16* __restrict__ c,
                               int n) {
  int i = blockIdx.x * blockDim.x + threadIdx.x;
  if (i < n) {
    const float gv = (float)g[i];
    const float s = 1.f / (1.f + __expf(-gv));
    g[i] = (__bf16)(s * (float)c[i]);
  }
}

// ---- elementwise: swiglu = silu(u1) * u2 (in-place into u1) ----------------
__global__ void swiglu_kernel(__bf16* __restrict__ u1, const __bf16* __restrict__ u2,
                              int n) {
  int i = blockIdx.x * blockDim.x + threadIdx.x;
  if (i < n) {
    const float a = (float)u1[i];
    const float s = a / (1.f + __expf(-a));
    u1[i] = (__bf16)(s * (float)u2[i]);
  }
}

extern "C" void kernel_launch(void* const* d_in, const int* in_sizes, int n_in,
                              void* d_out, int out_size, void* d_ws, size_t ws_size,
                              hipStream_t stream) {
  (void)in_sizes; (void)n_in; (void)out_size; (void)ws_size;
  const float* x     = (const float*)d_in[0];
  const float* rms_w = (const float*)d_in[1];
  const float* wq = (const float*)d_in[2];   const float* bq = (const float*)d_in[3];
  const float* wk = (const float*)d_in[4];   const float* bk = (const float*)d_in[5];
  const float* wv = (const float*)d_in[6];   const float* bv = (const float*)d_in[7];
  const float* wg = (const float*)d_in[8];   const float* bg = (const float*)d_in[9];
  const float* wo = (const float*)d_in[10];  const float* bo = (const float*)d_in[11];
  const float* w1 = (const float*)d_in[12];  const float* b1 = (const float*)d_in[13];
  const float* w2 = (const float*)d_in[14];  const float* b2 = (const float*)d_in[15];
  const float* wd = (const float*)d_in[16];  const float* bd = (const float*)d_in[17];
  float* out = (float*)d_out;

  const size_t NTOK = (size_t)B_ * S_;       // 4096 rows
  __bf16* ws   = (__bf16*)d_ws;
  __bf16* xn   = ws;                         // [NTOK, D]
  __bf16* q    = xn   + NTOK * D_;
  __bf16* k    = q    + NTOK * D_;
  __bf16* v    = k    + NTOK * D_;
  __bf16* gate = v    + NTOK * D_;
  __bf16* ctx  = gate + NTOK * D_;
  __bf16* u1   = ctx  + NTOK * D_;           // [NTOK, DFF]
  __bf16* u2   = u1   + NTOK * DFF_;
  __bf16* wqb  = u2   + NTOK * DFF_;         // bf16 weights
  __bf16* wkb  = wqb  + (size_t)D_ * D_;
  __bf16* wvb  = wkb  + (size_t)D_ * D_;
  __bf16* wgb  = wvb  + (size_t)D_ * D_;
  __bf16* wob  = wgb  + (size_t)D_ * D_;
  __bf16* w1b  = wob  + (size_t)D_ * D_;
  __bf16* w2b  = w1b  + (size_t)DFF_ * D_;
  __bf16* wdb  = w2b  + (size_t)DFF_ * D_;

  // weight conversion (once per launch)
  const int nDD  = D_ * D_;
  const int nFD  = DFF_ * D_;
  const int gDD  = nDD / 4 / 256;
  const int gFD  = nFD / 4 / 256;
  cvt_bf16_kernel<<<gDD, 256, 0, stream>>>(wq, wqb, nDD);
  cvt_bf16_kernel<<<gDD, 256, 0, stream>>>(wk, wkb, nDD);
  cvt_bf16_kernel<<<gDD, 256, 0, stream>>>(wv, wvb, nDD);
  cvt_bf16_kernel<<<gDD, 256, 0, stream>>>(wg, wgb, nDD);
  cvt_bf16_kernel<<<gDD, 256, 0, stream>>>(wo, wob, nDD);
  cvt_bf16_kernel<<<gFD, 256, 0, stream>>>(w1, w1b, nFD);
  cvt_bf16_kernel<<<gFD, 256, 0, stream>>>(w2, w2b, nFD);
  cvt_bf16_kernel<<<gFD, 256, 0, stream>>>(wd, wdb, nFD);

  rmsnorm_kernel<<<(int)NTOK, 256, 0, stream>>>(x, rms_w, xn);

  dim3 gD(D_ / 128, NTOK / 128);             // (8, 32)
  dim3 gF(DFF_ / 128, NTOK / 128);           // (32, 32)
  gemm_wmma<0><<<gD, 256, 0, stream>>>(xn, wqb, bq, q,    D_,   D_);
  gemm_wmma<0><<<gD, 256, 0, stream>>>(xn, wkb, bk, k,    D_,   D_);
  gemm_wmma<0><<<gD, 256, 0, stream>>>(xn, wvb, bv, v,    D_,   D_);
  gemm_wmma<0><<<gD, 256, 0, stream>>>(xn, wgb, bg, gate, D_,   D_);
  gemm_wmma<0><<<gF, 256, 0, stream>>>(xn, w1b, b1, u1,   DFF_, D_);
  gemm_wmma<0><<<gF, 256, 0, stream>>>(xn, w2b, b2, u2,   DFF_, D_);

  dim3 ga(S_ / 128, H_, B_);
  attention_wmma<<<ga, 256, 0, stream>>>(q, k, v, ctx);

  const int ne1 = (int)(NTOK * D_);
  gatemul_kernel<<<(ne1 + 255) / 256, 256, 0, stream>>>(gate, ctx, ne1);
  const int ne2 = (int)(NTOK * DFF_);
  swiglu_kernel<<<(ne2 + 255) / 256, 256, 0, stream>>>(u1, u2, ne2);

  gemm_wmma<1><<<gD, 256, 0, stream>>>(gate, wob, bo, out, D_, D_);    // write
  gemm_wmma<2><<<gD, 256, 0, stream>>>(u1,   wdb, bd, out, D_, DFF_);  // accumulate
}
